// AtomGMMProjector_38826504355895
// MI455X (gfx1250) — matrix operand order repaired and checked by reading, not compiled, not run on hardware
//
#include <hip/hip_runtime.h>

// ---------------------------------------------------------------------------
// AtomGMMProjector for MI455X (gfx1250, wave32, WMMA).
// out[b,y,x] = sum_n amp[n] * exp(-(line[y]-rc_y[b,n])^2/(2s^2))
//                          * exp(-(line[x]-rc_x[b,n])^2/(2s^2))
// Per batch: (256x4096)x(4096x256) GEMM -> v_wmma_f32_16x16x32_f16, with the
// f16 Gaussian operand tiles generated on the fly (v_exp_f32) into LDS,
// double-buffered and software-pipelined against the WMMA k-steps.
// ---------------------------------------------------------------------------

typedef _Float16 v16h __attribute__((ext_vector_type(16)));
typedef _Float16 v8h  __attribute__((ext_vector_type(8)));
typedef __fp16   h2   __attribute__((ext_vector_type(2)));   // cvt_pkrtz result type
typedef float    v8f  __attribute__((ext_vector_type(8)));

#define SIDE   256
#define NB     32
#define NATOM  4096
#define KC     128            // atoms per K-chunk
#define NCHUNK (NATOM / KC)   // 32
#define TILE   128            // output tile per workgroup
#define LDH    136            // LDS row stride in halves (128 + 8 pad = 16B pad)

// ---- stage per-atom params {rc_x, rc_y, log2e/(2 sigma^2), amp} ----
__device__ __forceinline__ void stage_params(
    int chunk, int b, int tid,
    const float* __restrict__ centers, const float* __restrict__ sigmas,
    const float* __restrict__ amps,
    float r00, float r01, float r02, float r10, float r11, float r12,
    float4* __restrict__ dst)
{
    if (tid < KC) {
        const int atom = chunk * KC + tid;
        const float* c = centers + ((size_t)b * NATOM + atom) * 3;
        const float c0 = c[0], c1 = c[1], c2 = c[2];
        const float sg = sigmas[atom];
        float4 p;
        p.x = r00 * c0 + r01 * c1 + r02 * c2;            // rc_x
        p.y = r10 * c0 + r11 * c1 + r12 * c2;            // rc_y
        p.z = 0.72134752044f / (sg * sg);                // log2(e)/(2 s^2)
        p.w = amps[atom];
        dst[tid] = p;
    }
}

// ---- one pipelined chunk: stage params(chunk+2), fill tiles(chunk+1),
//      WMMA(chunk); buffers resolved at compile time by the caller ----
template<bool STAGE, bool FILL>
__device__ __forceinline__ void chunk_body(
    int chunk, int b, int tid,
    const float* __restrict__ centers, const float* __restrict__ sigmas,
    const float* __restrict__ amps,
    float r00, float r01, float r02, float r10, float r11, float r12,
    float ly, float lx, int a0,
    float4* __restrict__ prmStage,            // dest for params(chunk+2)
    const float4* __restrict__ prmFill,       // params(chunk+1)
    unsigned int* __restrict__ nxtA32,        // fill dest ptrs (tiles chunk+1)
    unsigned int* __restrict__ nxtB32,
    const _Float16* __restrict__ curARow,     // per-lane A row ptr (tiles chunk)
    const _Float16* __restrict__ curB,        // base of current B tile
    int aOff, int bOff, int l16,
    v8f (&acc)[8])
{
    if (STAGE)
        stage_params(chunk + 2, b, tid, centers, sigmas, amps,
                     r00, r01, r02, r10, r11, r12, prmStage);

    #pragma unroll
    for (int i = 0; i < 4; ++i) {
        // ---- quarter of the exp-fill for chunk+1, fused for co-execution ----
        if (FILL) {
            #pragma unroll
            for (int a2 = 0; a2 < 16; a2 += 2) {
                const int ai = i * 16 + a2;
                const float4 p0 = prmFill[a0 + ai];
                const float4 p1 = prmFill[a0 + ai + 1];
                const float dy0 = ly - p0.y, dy1 = ly - p1.y;
                const float dx0 = lx - p0.x, dx1 = lx - p1.x;
                const float va0 = p0.w * __builtin_amdgcn_exp2f(-(dy0 * dy0) * p0.z);
                const float va1 = p1.w * __builtin_amdgcn_exp2f(-(dy1 * dy1) * p1.z);
                const float vb0 = __builtin_amdgcn_exp2f(-(dx0 * dx0) * p0.z);
                const float vb1 = __builtin_amdgcn_exp2f(-(dx1 * dx1) * p1.z);
                const h2 pa = __builtin_amdgcn_cvt_pkrtz(va0, va1);
                const h2 pb = __builtin_amdgcn_cvt_pkrtz(vb0, vb1);
                nxtA32[ai >> 1] = __builtin_bit_cast(unsigned int, pa);
                nxtB32[ai >> 1] = __builtin_bit_cast(unsigned int, pb);
            }
        }
        // ---- WMMA k-step i: batch all fragment loads, then 8 WMMAs ----
        const int kh = i * 32;
        union { v16h v; v8h h[2]; } af;
        af.h[0] = *(const v8h*)(curARow + kh + aOff);
        af.h[1] = *(const v8h*)(curARow + kh + 16 + aOff);
        union { v16h v; v8h h[2]; } bf[8];
        #pragma unroll
        for (int ct = 0; ct < 8; ++ct) {
            const _Float16* bRow = curB + (ct * 16 + l16) * LDH;
            bf[ct].h[0] = *(const v8h*)(bRow + kh + bOff);
            bf[ct].h[1] = *(const v8h*)(bRow + kh + 16 + bOff);
        }
        #pragma unroll
        for (int ct = 0; ct < 8; ++ct) {
            acc[ct] = __builtin_amdgcn_wmma_f32_16x16x32_f16(
                false, af.v, false, bf[ct].v, (short)0, acc[ct], false, false);
        }
    }
    __syncthreads();
}

__global__ __launch_bounds__(256)
void atom_gmm_proj_kernel(const float* __restrict__ line,
                          const float* __restrict__ rot,
                          const float* __restrict__ centers,
                          const float* __restrict__ sigmas,
                          const float* __restrict__ amps,
                          float* __restrict__ out)
{
    // Double-buffered operand tiles (distinct arrays -> clean alias analysis).
    __shared__ _Float16 shA0[TILE * LDH];            // 34816 B each
    __shared__ _Float16 shA1[TILE * LDH];
    __shared__ _Float16 shB0[TILE * LDH];
    __shared__ _Float16 shB1[TILE * LDH];
    __shared__ float4   shP0[KC];                    // params, double-buffered
    __shared__ float4   shP1[KC];
    __shared__ float    shLine[SIDE];

    const int tid = threadIdx.x;
    const int b   = blockIdx.y;
    const int ty  = blockIdx.x >> 1;
    const int tx  = blockIdx.x & 1;
    const int rowBase = ty * TILE;
    const int colBase = tx * TILE;

    shLine[tid] = line[tid];                         // 256 threads == SIDE

    const float r00 = rot[b * 9 + 0], r01 = rot[b * 9 + 1], r02 = rot[b * 9 + 2];
    const float r10 = rot[b * 9 + 3], r11 = rot[b * 9 + 4], r12 = rot[b * 9 + 5];

    // fill roles: 2 threads per row, 64 atoms (one segment) each
    const int row = tid >> 1;
    const int seg = tid & 1;
    const int a0  = seg * 64;
    unsigned int* pA0 = (unsigned int*)shA0 + row * (LDH / 2) + seg * 32;
    unsigned int* pA1 = (unsigned int*)shA1 + row * (LDH / 2) + seg * 32;
    unsigned int* pB0 = (unsigned int*)shB0 + row * (LDH / 2) + seg * 32;
    unsigned int* pB1 = (unsigned int*)shB1 + row * (LDH / 2) + seg * 32;

    // WMMA roles: wave w owns rows [16w,16w+16), all 8 column tiles
    const int  lane = tid & 31;
    const int  wave = tid >> 5;
    const int  l16  = lane & 15;
    const bool hiHalf = (lane >= 16);
    const _Float16* aRow0 = shA0 + (wave * 16 + l16) * LDH;
    const _Float16* aRow1 = shA1 + (wave * 16 + l16) * LDH;
    const int aOff = hiHalf ? 8 : 0;                 // A frag K interleave
    const int bOff = hiHalf ? 8 : 0;                 // B frag K interleave

    // ---- prologue: params(0); then params(1) + full fill of tiles(0) ----
    stage_params(0, b, tid, centers, sigmas, amps, r00, r01, r02, r10, r11, r12, shP0);
    __syncthreads();
    stage_params(1, b, tid, centers, sigmas, amps, r00, r01, r02, r10, r11, r12, shP1);
    {
        const float ly = shLine[rowBase + row];
        const float lx = shLine[colBase + row];
        #pragma unroll 8
        for (int a = 0; a < 64; a += 2) {
            const float4 p0 = shP0[a0 + a];
            const float4 p1 = shP0[a0 + a + 1];
            const float dy0 = ly - p0.y, dy1 = ly - p1.y;
            const float dx0 = lx - p0.x, dx1 = lx - p1.x;
            const float va0 = p0.w * __builtin_amdgcn_exp2f(-(dy0 * dy0) * p0.z);
            const float va1 = p1.w * __builtin_amdgcn_exp2f(-(dy1 * dy1) * p1.z);
            const float vb0 = __builtin_amdgcn_exp2f(-(dx0 * dx0) * p0.z);
            const float vb1 = __builtin_amdgcn_exp2f(-(dx1 * dx1) * p1.z);
            const h2 pa = __builtin_amdgcn_cvt_pkrtz(va0, va1);
            const h2 pb = __builtin_amdgcn_cvt_pkrtz(vb0, vb1);
            pA0[a >> 1] = __builtin_bit_cast(unsigned int, pa);
            pB0[a >> 1] = __builtin_bit_cast(unsigned int, pb);
        }
    }
    __syncthreads();

    const float ly = shLine[rowBase + row];
    const float lx = shLine[colBase + row];

    v8f acc[8];
    #pragma unroll
    for (int i = 0; i < 8; ++i)
        #pragma unroll
        for (int j = 0; j < 8; ++j) acc[i][j] = 0.0f;

    // ---- pipelined main loop: chunks 0..29 in parity pairs ----
    for (int cc = 0; cc < NCHUNK - 2; cc += 2) {
        // even chunk: compute on buf0, fill buf1, params P(cc+1)=shP1 -> stage shP0
        chunk_body<true, true>(cc, b, tid, centers, sigmas, amps,
                               r00, r01, r02, r10, r11, r12, ly, lx, a0,
                               shP0, shP1, pA1, pB1, aRow0, shB0,
                               aOff, bOff, l16, acc);
        // odd chunk: compute on buf1, fill buf0, params P(cc+2)=shP0 -> stage shP1
        chunk_body<true, true>(cc + 1, b, tid, centers, sigmas, amps,
                               r00, r01, r02, r10, r11, r12, ly, lx, a0,
                               shP1, shP0, pA0, pB0, aRow1, shB1,
                               aOff, bOff, l16, acc);
    }
    // chunk 30: compute buf0, fill tiles(31) into buf1, no more staging
    chunk_body<false, true>(NCHUNK - 2, b, tid, centers, sigmas, amps,
                            r00, r01, r02, r10, r11, r12, ly, lx, a0,
                            shP0, shP1, pA1, pB1, aRow0, shB0,
                            aOff, bOff, l16, acc);
    // chunk 31: compute buf1 only
    chunk_body<false, false>(NCHUNK - 1, b, tid, centers, sigmas, amps,
                             r00, r01, r02, r10, r11, r12, ly, lx, a0,
                             shP1, shP0, pA0, pB0, aRow1, shB1,
                             aOff, bOff, l16, acc);

    // ---- write D: VGPR j -> row (j or j+8 by lane half), col = lane%16 ----
    const int mBase = rowBase + wave * 16 + (hiHalf ? 8 : 0);
    float* outB = out + (size_t)b * SIDE * SIDE;
    #pragma unroll
    for (int ct = 0; ct < 8; ++ct) {
        const int col = colBase + ct * 16 + l16;
        #pragma unroll
        for (int j = 0; j < 8; ++j) {
            outB[(size_t)(mBase + j) * SIDE + col] = acc[ct][j];
        }
    }
}

extern "C" void kernel_launch(void* const* d_in, const int* in_sizes, int n_in,
                              void* d_out, int out_size, void* d_ws, size_t ws_size,
                              hipStream_t stream) {
    (void)in_sizes; (void)n_in; (void)d_ws; (void)ws_size; (void)out_size;
    const float* line = (const float*)d_in[0];   // (256,)
    const float* rot  = (const float*)d_in[1];   // (32,3,3)
    const float* cent = (const float*)d_in[2];   // (32,4096,3)
    const float* sig  = (const float*)d_in[3];   // (4096,)
    const float* amp  = (const float*)d_in[4];   // (4096,)
    float* out = (float*)d_out;                  // (32,256,256)

    dim3 grid(4, NB);                            // 4 tiles of 128x128 per batch
    atom_gmm_proj_kernel<<<grid, 256, 0, stream>>>(line, rot, cent, sig, amp, out);
}
